// SolverInTheLoop_27728308863254
// MI455X (gfx1250) — matrix operand order repaired and checked by reading, not compiled, not run on hardware
//
#include <hip/hip_runtime.h>

// ---------------------------------------------------------------------------
// SolverInTheLoop for MI455X (gfx1250): SPH + GNS GNN.
// v3: fully fused 3-layer MLP kernel. Per 128-row block (4 waves), each wave
//     owns a 32-row strip; layer outputs stay in a wave-private LDS strip so
//     MLP intermediates never touch HBM. W^T staged via double-buffered async
//     global->LDS chunks. bf16 WMMA (16x16x32), f32 accumulate.
// ---------------------------------------------------------------------------

typedef __attribute__((ext_vector_type(16))) __bf16 v16bf;
typedef __attribute__((ext_vector_type(8)))  __bf16 v8bf;
typedef __attribute__((ext_vector_type(8)))  float  v8f;

#define HD __device__ __forceinline__

// ---- problem constants -----------------------------------------------------
constexpr int   NPART = 32 * 32 * 16;   // 16384
constexpr int   DEG   = 10;
constexpr int   NEDGE = NPART * DEG;    // 163840
constexpr float Hk    = 0.2f;
constexpr float SIGMA = 3.0f / (359.0f * 3.14159265358979f * (0.2f * 0.2f * 0.2f));
constexpr float DT    = 0.01f;
constexpr float STIL  = DT / 2.0f;      // DT / NSITL
constexpr float MASS  = 0.2f * 0.2f * 0.2f; // DX^3
constexpr float ETA0  = 0.01f;
constexpr float P_REF = 100.0f;
constexpr float P_BG  = 0.02f * 100.0f;
constexpr float GNN_R = 0.6f;

HD float p4(float x) { float x2 = x * x; return x2 * x2; }
HD float p5(float x) { float x2 = x * x; return x2 * x2 * x; }
HD __bf16 tobf(float f) { return (__bf16)f; }
HD float  tof(__bf16 b) { return (float)b; }

HD v16bf cat16(v8bf lo, v8bf hi) {
    v16bf r;
#pragma unroll
    for (int i = 0; i < 8; ++i) { r[i] = lo[i]; r[i + 8] = hi[i]; }
    return r;
}

// ---------------------------------------------------------------------------
// Fused MLP (2 or 3 layers of width 128):
//   h1 = relu(A @ W1 + b1);  h2 = relu(h1 @ W2 + b2);
//   out = h2 @ W3 + b3 (+resid)            [3-layer, W3t != null]
//   out = h2                               [2-layer, W3t == null]
// A[M x K1tot] = concat of up to 3 segments (each a multiple of 32 wide),
// segments 1/2 optionally row-gathered. Intermediates live in a wave-private
// 32x128 bf16 LDS strip (no HBM round trips, no extra barriers).
// 128 threads = 4 waves; wave = 2 M-tiles x 8 N-tiles = 16 WMMA / K-chunk.
// Requires M % 128 == 0.
// ---------------------------------------------------------------------------
__global__ void __launch_bounds__(128)
sitl_fused_mlp(const __bf16* __restrict__ s0,
               const __bf16* __restrict__ s1,
               const __bf16* __restrict__ s2,
               const int* __restrict__ i1, const int* __restrict__ i2,
               int K0, int K1, int K2,
               const __bf16* __restrict__ W1t, const float* __restrict__ b1,
               const __bf16* __restrict__ W2t, const float* __restrict__ b2,
               const __bf16* __restrict__ W3t, const float* __restrict__ b3,
               const __bf16* __restrict__ resid,
               __bf16* __restrict__ out_bf,
               int M)
{
    __shared__ __bf16 actS[4 * 32 * 128];        // 32KB: per-wave 32x128 strip
    __shared__ __bf16 ldsW[2 * 128 * 32];        // 16KB: W^T chunk dbl-buffer

    const int tid  = threadIdx.x;
    const int wave = tid >> 5;
    const int lane = tid & 31;
    const int l15  = lane & 15;
    const int hi   = lane >> 4;                  // K-half select (CDNA5 layout)

    const long row0  = ((long)blockIdx.x * 4 + wave) * 32;  // 2 M-tiles / wave
    const long arowA = row0 + l15;
    const long arowB = row0 + 16 + l15;
    __bf16* actW = actS + wave * (32 * 128);     // wave-private strip

    // per-lane segment base pointers for layer-1 A (gather resolved once)
    const __bf16* s0a = s0 + arowA * K0;
    const __bf16* s0b = s0 + arowB * K0;
    const __bf16* s1a = nullptr; const __bf16* s1b = nullptr;
    const __bf16* s2a = nullptr; const __bf16* s2b = nullptr;
    if (s1) {
        long ra = i1 ? (long)i1[arowA] : arowA;
        long rb = i1 ? (long)i1[arowB] : arowB;
        s1a = s1 + ra * K1; s1b = s1 + rb * K1;
    }
    if (s2) {
        long ra = i2 ? (long)i2[arowA] : arowA;
        long rb = i2 ? (long)i2[arowB] : arowB;
        s2a = s2 + ra * K2; s2b = s2 + rb * K2;
    }

    v8f c0[8], c1[8];

    auto zeroC = [&] {
#pragma unroll
        for (int t = 0; t < 8; ++t)
#pragma unroll
            for (int i = 0; i < 8; ++i) { c0[t][i] = 0.0f; c1[t][i] = 0.0f; }
    };

    // ---- async stage of one 128x32 W^T chunk into LDS buffer ---------------
    auto stageW = [&](const __bf16* Wt, int Ktot, int k0, int buf) {
#pragma unroll
        for (int s = 0; s < 4; ++s) {            // 512 x 16B per chunk
            int v  = tid + s * 128;
            int n  = v >> 2;
            int kk = (v & 3) * 8;
            const __bf16* src = Wt + (long)n * Ktot + k0 + kk;
            unsigned int  dst = (unsigned int)(unsigned long long)
                                (ldsW + buf * (128 * 32) + v * 8);
            unsigned long long ga = (unsigned long long)src;
            asm volatile("global_load_async_to_lds_b128 %0, %1, off"
                         :: "v"(dst), "v"(ga) : "memory");
        }
    };
    auto waitAsync = [&] {
        asm volatile("s_wait_asynccnt 0x0" ::: "memory");
    };

    // ---- one GEMM phase: c += A @ W, A from global concat (mode 0) or the
    //      wave-private LDS strip (mode 1) ------------------------------------
    auto runPhase = [&](const __bf16* Wt, int Ktot, int mode) {
        const int nch = Ktot >> 5;
        stageW(Wt, Ktot, 0, 0);
        waitAsync();
        __syncthreads();
        for (int ci = 0; ci < nch; ++ci) {
            const int k0  = ci << 5;
            const int buf = ci & 1;
            if (ci + 1 < nch) stageW(Wt, Ktot, k0 + 32, 1 - buf);

            v16bf a0, a1;
            if (mode == 0) {
                // segment select is UNIFORM per chunk (segments %32 == 0)
                const __bf16 *pa, *pb;
                if (k0 < K0)           { pa = s0a + k0;             pb = s0b + k0; }
                else if (k0 < K0 + K1) { pa = s1a + (k0 - K0);      pb = s1b + (k0 - K0); }
                else                   { pa = s2a + (k0 - K0 - K1); pb = s2b + (k0 - K0 - K1); }
                a0 = cat16(*(const v8bf*)(pa + hi * 8),
                           *(const v8bf*)(pa + hi * 8 + 16));
                a1 = cat16(*(const v8bf*)(pb + hi * 8),
                           *(const v8bf*)(pb + hi * 8 + 16));
            } else {
                const __bf16* pa = actW + l15 * 128 + k0 + hi * 8;
                const __bf16* pb = actW + (16 + l15) * 128 + k0 + hi * 8;
                a0 = cat16(*(const v8bf*)pa, *(const v8bf*)(pa + 16));
                a1 = cat16(*(const v8bf*)pb, *(const v8bf*)(pb + 16));
            }

            const __bf16* wb = ldsW + buf * (128 * 32);
#pragma unroll
            for (int t = 0; t < 8; ++t) {
                const __bf16* bp = wb + (t * 16 + l15) * 32 + hi * 8;
                v16bf b = cat16(*(const v8bf*)bp, *(const v8bf*)(bp + 16));
                c0[t] = __builtin_amdgcn_wmma_f32_16x16x32_bf16(
                            false, a0, false, b, (short)0, c0[t], false, false);
                c1[t] = __builtin_amdgcn_wmma_f32_16x16x32_bf16(
                            false, a1, false, b, (short)0, c1[t], false, false);
            }
            waitAsync();
            __syncthreads();
        }
    };

    // ---- bias+relu accumulators -> wave-private LDS strip (wave-local) -----
    auto storeAct = [&](const float* bias) {
#pragma unroll
        for (int g = 0; g < 2; ++g) {
            const int lbase = g * 16 + hi * 8;
#pragma unroll
            for (int t = 0; t < 8; ++t) {
                int col = t * 16 + l15;
                float bv = bias[col];
#pragma unroll
                for (int rI = 0; rI < 8; ++rI) {
                    float v = (g == 0 ? c0[t][rI] : c1[t][rI]) + bv;
                    v = v > 0.0f ? v : 0.0f;     // hidden layers always relu
                    actW[(lbase + rI) * 128 + col] = tobf(v);
                }
            }
        }
    };

    // ---- final epilogue: bias (+resid) (+relu) -> global bf16 --------------
    auto storeOut = [&](const float* bias, int relu) {
#pragma unroll
        for (int g = 0; g < 2; ++g) {
            const long obase = row0 + g * 16 + hi * 8;
#pragma unroll
            for (int t = 0; t < 8; ++t) {
                int col = t * 16 + l15;
                float bv = bias[col];
#pragma unroll
                for (int rI = 0; rI < 8; ++rI) {
                    long orow = obase + rI;
                    float v = (g == 0 ? c0[t][rI] : c1[t][rI]) + bv;
                    if (resid) v += tof(resid[orow * 128 + col]);
                    if (relu)  v = v > 0.0f ? v : 0.0f;
                    out_bf[orow * 128 + col] = tobf(v);
                }
            }
        }
    };

    // ---- layer 1 -----------------------------------------------------------
    zeroC();
    runPhase(W1t, K0 + K1 + K2, 0);
    storeAct(b1);
    // ---- layer 2 -----------------------------------------------------------
    zeroC();
    runPhase(W2t, 128, 1);
    if (W3t) {
        storeAct(b2);
        // ---- layer 3 -------------------------------------------------------
        zeroC();
        runPhase(W3t, 128, 1);
        storeOut(b3, 0);
    } else {
        storeOut(b2, 1);                 // 2-layer mode (decoder trunk)
    }
}

// ---------------------------------------------------------------------------
// Weight prep: f32 W[K x 128] (row-major) -> bf16 W^T[128 x Kpad], zero padded
// ---------------------------------------------------------------------------
__global__ void sitl_prep_w(const float* __restrict__ W, __bf16* __restrict__ Wt,
                            int K, int Kpad)
{
    int idx = blockIdx.x * blockDim.x + threadIdx.x;
    if (idx >= 128 * Kpad) return;
    int n = idx / Kpad, k = idx % Kpad;
    Wt[idx] = (k < K) ? tobf(W[k * 128 + n]) : tobf(0.0f);
}

// ---------------------------------------------------------------------------
// Init / SPH / features / segment-sum / decoder / integrate / final
// ---------------------------------------------------------------------------
__global__ void sitl_init(const float* __restrict__ abs_pos,
                          const float* __restrict__ vel_hist,
                          float* r, float* u, float* r0, float* u0)
{
    int idx = blockIdx.x * blockDim.x + threadIdx.x;
    if (idx >= NPART * 3) return;
    float rr = abs_pos[idx];
    float uu = vel_hist[idx] / DT;     // (v*STD+MEAN)/DT, STD=1 MEAN=0
    r[idx] = rr; u[idx] = uu; r0[idx] = rr; u0[idx] = uu;
}

// rho + p per node: receiver segments are contiguous runs of DEG edges
__global__ void sitl_sph_rho(const float* __restrict__ r,
                             const int* __restrict__ js,
                             float* __restrict__ rho, float* __restrict__ p)
{
    int n = blockIdx.x * blockDim.x + threadIdx.x;
    if (n >= NPART) return;
    float rx = r[n*3+0], ry = r[n*3+1], rz = r[n*3+2];
    float s = 0.0f;
#pragma unroll
    for (int d = 0; d < DEG; ++d) {
        int j = js[n * DEG + d];
        float dx = rx - r[j*3+0], dy = ry - r[j*3+1], dz = rz - r[j*3+2];
        float dd = sqrtf(fmaxf(dx*dx + dy*dy + dz*dz, 1e-16f));
        float q  = dd / Hk;
        s += SIGMA * (p5(fmaxf(3.0f-q,0.0f)) - 6.0f*p5(fmaxf(2.0f-q,0.0f))
                      + 15.0f*p5(fmaxf(1.0f-q,0.0f)));
    }
    float rh = MASS * s;
    rho[n] = rh;
    p[n] = P_REF * (rh - 1.0f) + P_BG;   // GAMMA == 1
}

__global__ void sitl_sph_acc(const float* __restrict__ r, const float* __restrict__ u,
                             const int* __restrict__ js,
                             const float* __restrict__ rho, const float* __restrict__ p,
                             float* __restrict__ acc)
{
    int n = blockIdx.x * blockDim.x + threadIdx.x;
    if (n >= NPART) return;
    float rx = r[n*3+0], ry = r[n*3+1], rz = r[n*3+2];
    float ux = u[n*3+0], uy = u[n*3+1], uz = u[n*3+2];
    float ri = rho[n], pi = p[n];
    float wi2 = (MASS / ri) * (MASS / ri);
    float eta_ij = 2.0f * ETA0 * ETA0 / (2.0f * ETA0 + 1e-8f);
    float ax = 0.0f, ay = 0.0f, az = 0.0f;
#pragma unroll
    for (int d = 0; d < DEG; ++d) {
        int j = js[n * DEG + d];
        float dx = rx - r[j*3+0], dy = ry - r[j*3+1], dz = rz - r[j*3+2];
        float dd = sqrtf(fmaxf(dx*dx + dy*dy + dz*dz, 1e-16f));
        float q  = dd / Hk;
        float gw = (SIGMA / Hk) * (-5.0f*p4(fmaxf(3.0f-q,0.0f))
                                   + 30.0f*p4(fmaxf(2.0f-q,0.0f))
                                   - 75.0f*p4(fmaxf(1.0f-q,0.0f)));
        float rj = rho[j], pj = p[j];
        float p_ij = (rj * pi + ri * pj) / (ri + rj);
        float wv = (wi2 + (MASS / rj) * (MASS / rj)) / MASS;
        float c  = wv * gw / (dd + 1e-8f);
        ax += c * (-p_ij * dx + eta_ij * (ux - u[j*3+0]));
        ay += c * (-p_ij * dy + eta_ij * (uy - u[j*3+1]));
        az += c * (-p_ij * dz + eta_ij * (uz - u[j*3+2]));
    }
    acc[n*3+0] = ax; acc[n*3+1] = ay; acc[n*3+2] = az;
}

// node feature: [u*DT (3), type_emb[tag] (16), pad] -> bf16 N x 32
__global__ void sitl_node_feat(const float* __restrict__ u, const int* __restrict__ tag,
                               const float* __restrict__ emb, __bf16* __restrict__ out)
{
    int n = blockIdx.x * blockDim.x + threadIdx.x;
    if (n >= NPART) return;
    __bf16 t[32];
    int tg = tag[n];
#pragma unroll
    for (int c = 0; c < 32; ++c) {
        float v = 0.0f;
        if (c < 3)       v = u[n*3+c] * DT;
        else if (c < 19) v = emb[tg * 16 + (c - 3)];
        t[c] = tobf(v);
    }
    v8bf* dst = (v8bf*)(out + (long)n * 32);
#pragma unroll
    for (int k = 0; k < 4; ++k) dst[k] = ((v8bf*)t)[k];
}

// edge feature: [rel_disp (3), rel_dist (1), pad] -> bf16 E x 32
__global__ void sitl_edge_feat(const float* __restrict__ r,
                               const int* __restrict__ recv, const int* __restrict__ send,
                               __bf16* __restrict__ out)
{
    int e = blockIdx.x * blockDim.x + threadIdx.x;
    if (e >= NEDGE) return;
    int i = recv[e], j = send[e];
    float dx = (r[i*3+0]-r[j*3+0]) / GNN_R;
    float dy = (r[i*3+1]-r[j*3+1]) / GNN_R;
    float dz = (r[i*3+2]-r[j*3+2]) / GNN_R;
    float dd = sqrtf(fmaxf(dx*dx+dy*dy+dz*dz, 1e-16f));
    __bf16 t[32];
    t[0]=tobf(dx); t[1]=tobf(dy); t[2]=tobf(dz); t[3]=tobf(dd);
#pragma unroll
    for (int c = 4; c < 32; ++c) t[c] = tobf(0.0f);
    v8bf* dst = (v8bf*)(out + (long)e * 32);
#pragma unroll
    for (int k = 0; k < 4; ++k) dst[k] = ((v8bf*)t)[k];
}

// agg[n,c] = sum_{d<DEG} edge[n*DEG+d, c]   (contiguous receiver segments,
// lanes cover consecutive c -> fully coalesced, no atomics)
__global__ void sitl_agg(const __bf16* __restrict__ edge, __bf16* __restrict__ agg)
{
    long idx = (long)blockIdx.x * blockDim.x + threadIdx.x;
    if (idx >= (long)NPART * 128) return;
    long n = idx >> 7; int c = (int)(idx & 127);
    const __bf16* base = edge + (n * DEG) * 128 + c;
    float s = 0.0f;
#pragma unroll
    for (int d = 0; d < DEG; ++d) s += tof(base[d * 128]);
    agg[idx] = tobf(s);
}

// decoder layer3: bf16 [N x 128] @ f32 [128 x 3] + b3 -> f32 [N x 3]
__global__ void sitl_dec3(const __bf16* __restrict__ h, const float* __restrict__ w3,
                          const float* __restrict__ b3, float* __restrict__ out)
{
    int n = blockIdx.x * blockDim.x + threadIdx.x;
    if (n >= NPART) return;
    float a0 = b3[0], a1 = b3[1], a2 = b3[2];
    const __bf16* hr = h + (long)n * 128;
#pragma unroll 4
    for (int k = 0; k < 128; ++k) {
        float hv = tof(hr[k]);
        a0 += hv * w3[k*3+0];
        a1 += hv * w3[k*3+1];
        a2 += hv * w3[k*3+2];
    }
    out[n*3+0]=a0; out[n*3+1]=a1; out[n*3+2]=a2;
}

__global__ void sitl_integrate(float* __restrict__ u, float* __restrict__ r,
                               const float* __restrict__ acc_sph,
                               const float* __restrict__ gns)
{
    int idx = blockIdx.x * blockDim.x + threadIdx.x;
    if (idx >= NPART * 3) return;
    float acc_gns = gns[idx] / (DT * DT);
    float uu = u[idx] + STIL * (acc_sph[idx] + acc_gns);
    u[idx] = uu;
    r[idx] = r[idx] + STIL * uu;
}

__global__ void sitl_final(const float* __restrict__ r, const float* __restrict__ r0,
                           const float* __restrict__ u0, float* __restrict__ out)
{
    int idx = blockIdx.x * blockDim.x + threadIdx.x;
    if (idx >= NPART * 3) return;
    out[idx] = r[idx] - r0[idx] - u0[idx] * DT;   // normalization is identity
}

// ---------------------------------------------------------------------------
// Host side
// ---------------------------------------------------------------------------
static inline void mlp3(hipStream_t st, int M,
                        const __bf16* s0, int K0,
                        const __bf16* s1, int K1, const int* i1,
                        const __bf16* s2, int K2, const int* i2,
                        const __bf16* W1t, const float* b1,
                        const __bf16* W2t, const float* b2,
                        const __bf16* W3t, const float* b3,
                        const __bf16* resid, __bf16* outb)
{
    sitl_fused_mlp<<<dim3(M / 128), dim3(128), 0, st>>>(
        s0, s1, s2, i1, i2, K0, K1, K2, W1t, b1, W2t, b2, W3t, b3,
        resid, outb, M);
}

extern "C" void kernel_launch(void* const* d_in, const int* in_sizes, int n_in,
                              void* d_out, int out_size, void* d_ws, size_t ws_size,
                              hipStream_t stream)
{
    // ---- input layout (setup_inputs dict order, params tree-sorted) --------
    const float* abs_pos  = (const float*)d_in[0];
    const float* vel_hist = (const float*)d_in[1];
    const int*   tag      = (const int*)d_in[2];
    const int*   sph_i    = (const int*)d_in[3];  (void)sph_i;
    const int*   sph_j    = (const int*)d_in[4];
    const int*   gnn_recv = (const int*)d_in[5];
    const int*   gnn_send = (const int*)d_in[6];
    const float* dec_b1 = (const float*)d_in[7];
    const float* dec_b2 = (const float*)d_in[8];
    const float* dec_b3 = (const float*)d_in[9];
    const float* dec_w1 = (const float*)d_in[10];
    const float* dec_w2 = (const float*)d_in[11];
    const float* dec_w3 = (const float*)d_in[12];
    const float* ee_b1  = (const float*)d_in[13];
    const float* ee_b2  = (const float*)d_in[14];
    const float* ee_b3  = (const float*)d_in[15];
    const float* ee_w1  = (const float*)d_in[16];
    const float* ee_w2  = (const float*)d_in[17];
    const float* ee_w3  = (const float*)d_in[18];
    const float* eu_b1  = (const float*)d_in[19];   // (3,128)
    const float* eu_b2  = (const float*)d_in[20];
    const float* eu_b3  = (const float*)d_in[21];
    const float* eu_w1  = (const float*)d_in[22];   // (3,384,128)
    const float* eu_w2  = (const float*)d_in[23];   // (3,128,128)
    const float* eu_w3  = (const float*)d_in[24];
    const float* ne_b1  = (const float*)d_in[25];
    const float* ne_b2  = (const float*)d_in[26];
    const float* ne_b3  = (const float*)d_in[27];
    const float* ne_w1  = (const float*)d_in[28];   // (19,128)
    const float* ne_w2  = (const float*)d_in[29];
    const float* ne_w3  = (const float*)d_in[30];
    const float* nu_b1  = (const float*)d_in[31];   // (3,128)
    const float* nu_b2  = (const float*)d_in[32];
    const float* nu_b3  = (const float*)d_in[33];
    const float* nu_w1  = (const float*)d_in[34];   // (3,256,128)
    const float* nu_w2  = (const float*)d_in[35];
    const float* nu_w3  = (const float*)d_in[36];
    const float* type_emb = (const float*)d_in[37];
    (void)in_sizes; (void)n_in; (void)out_size; (void)ws_size;

    // ---- workspace bump allocator ------------------------------------------
    size_t off = 0;
    auto alloc = [&](size_t bytes) -> void* {
        void* p = (char*)d_ws + off;
        off = (off + bytes + 255) & ~(size_t)255;
        return p;
    };
    const long N = NPART, E = NEDGE;
    float* r       = (float*)alloc(N * 3 * 4);
    float* u       = (float*)alloc(N * 3 * 4);
    float* r0      = (float*)alloc(N * 3 * 4);
    float* u0      = (float*)alloc(N * 3 * 4);
    float* rho     = (float*)alloc(N * 4);
    float* pbuf    = (float*)alloc(N * 4);
    float* acc_sph = (float*)alloc(N * 3 * 4);
    float* gns     = (float*)alloc(N * 3 * 4);
    __bf16* node_in = (__bf16*)alloc(N * 32 * 2);
    __bf16* edge_in = (__bf16*)alloc(E * 32 * 2);
    __bf16* node    = (__bf16*)alloc(N * 128 * 2);
    __bf16* agg_bf  = (__bf16*)alloc(N * 128 * 2);
    __bf16* dec_h   = (__bf16*)alloc(N * 128 * 2);
    __bf16* edge    = (__bf16*)alloc(E * 128 * 2);
    // transposed bf16 weights
    __bf16* t_ne1 = (__bf16*)alloc(128 * 32 * 2);
    __bf16* t_ne2 = (__bf16*)alloc(128 * 128 * 2);
    __bf16* t_ne3 = (__bf16*)alloc(128 * 128 * 2);
    __bf16* t_ee1 = (__bf16*)alloc(128 * 32 * 2);
    __bf16* t_ee2 = (__bf16*)alloc(128 * 128 * 2);
    __bf16* t_ee3 = (__bf16*)alloc(128 * 128 * 2);
    __bf16* t_eu1[3]; __bf16* t_eu2[3]; __bf16* t_eu3[3];
    __bf16* t_nu1[3]; __bf16* t_nu2[3]; __bf16* t_nu3[3];
    for (int m = 0; m < 3; ++m) {
        t_eu1[m] = (__bf16*)alloc(128 * 384 * 2);
        t_eu2[m] = (__bf16*)alloc(128 * 128 * 2);
        t_eu3[m] = (__bf16*)alloc(128 * 128 * 2);
        t_nu1[m] = (__bf16*)alloc(128 * 256 * 2);
        t_nu2[m] = (__bf16*)alloc(128 * 128 * 2);
        t_nu3[m] = (__bf16*)alloc(128 * 128 * 2);
    }
    __bf16* t_de1 = (__bf16*)alloc(128 * 128 * 2);
    __bf16* t_de2 = (__bf16*)alloc(128 * 128 * 2);

    auto prep = [&](const float* W, __bf16* Wt, int K, int Kpad) {
        int tot = 128 * Kpad;
        sitl_prep_w<<<dim3((tot + 255) / 256), dim3(256), 0, stream>>>(W, Wt, K, Kpad);
    };

    // ---- weight prep -------------------------------------------------------
    prep(ne_w1, t_ne1, 19, 32);  prep(ne_w2, t_ne2, 128, 128); prep(ne_w3, t_ne3, 128, 128);
    prep(ee_w1, t_ee1, 4, 32);   prep(ee_w2, t_ee2, 128, 128); prep(ee_w3, t_ee3, 128, 128);
    for (int m = 0; m < 3; ++m) {
        prep(eu_w1 + (long)m * 384 * 128, t_eu1[m], 384, 384);
        prep(eu_w2 + (long)m * 128 * 128, t_eu2[m], 128, 128);
        prep(eu_w3 + (long)m * 128 * 128, t_eu3[m], 128, 128);
        prep(nu_w1 + (long)m * 256 * 128, t_nu1[m], 256, 256);
        prep(nu_w2 + (long)m * 128 * 128, t_nu2[m], 128, 128);
        prep(nu_w3 + (long)m * 128 * 128, t_nu3[m], 128, 128);
    }
    prep(dec_w1, t_de1, 128, 128); prep(dec_w2, t_de2, 128, 128);

    // ---- init state --------------------------------------------------------
    const dim3 B256(256);
    const dim3 gE((unsigned)((E + 255) / 256));
    const dim3 gN((unsigned)((N + 255) / 256));
    const dim3 gN3((unsigned)((N * 3 + 255) / 256));
    const dim3 gN128((unsigned)((N * 128 + 255) / 256));

    sitl_init<<<gN3, B256, 0, stream>>>(abs_pos, vel_hist, r, u, r0, u0);

    for (int it = 0; it < 2; ++it) {
        // ---- SPH (atomic-free: contiguous DEG receiver segments) -----------
        sitl_sph_rho<<<gN, B256, 0, stream>>>(r, sph_j, rho, pbuf);
        sitl_sph_acc<<<gN, B256, 0, stream>>>(r, u, sph_j, rho, pbuf, acc_sph);

        // ---- GNN features --------------------------------------------------
        sitl_node_feat<<<gN, B256, 0, stream>>>(u, tag, type_emb, node_in);
        sitl_edge_feat<<<gE, B256, 0, stream>>>(r, gnn_recv, gnn_send, edge_in);

        // node encoder (fused 3-layer, M = N)
        mlp3(stream, N, node_in, 32, nullptr, 0, nullptr, nullptr, 0, nullptr,
             t_ne1, ne_b1, t_ne2, ne_b2, t_ne3, ne_b3, nullptr, node);
        // edge encoder (fused 3-layer, M = E)
        mlp3(stream, E, edge_in, 32, nullptr, 0, nullptr, nullptr, 0, nullptr,
             t_ee1, ee_b1, t_ee2, ee_b2, t_ee3, ee_b3, nullptr, edge);

        // ---- message passing ----------------------------------------------
        for (int m = 0; m < 3; ++m) {
            // edge update: A = concat(edge, node[recv], node[send]), +resid
            mlp3(stream, E, edge, 128, node, 128, gnn_recv, node, 128, gnn_send,
                 t_eu1[m], eu_b1 + m * 128, t_eu2[m], eu_b2 + m * 128,
                 t_eu3[m], eu_b3 + m * 128, edge, edge);

            // agg = segment_sum(edge, recv)  (contiguous segments, no atomics)
            sitl_agg<<<gN128, B256, 0, stream>>>(edge, agg_bf);

            // node update: A = concat(node, agg), +resid
            mlp3(stream, N, node, 128, agg_bf, 128, nullptr, nullptr, 0, nullptr,
                 t_nu1[m], nu_b1 + m * 128, t_nu2[m], nu_b2 + m * 128,
                 t_nu3[m], nu_b3 + m * 128, node, node);
        }

        // ---- decoder (fused 2-layer trunk + tiny 128->3 layer) -------------
        mlp3(stream, N, node, 128, nullptr, 0, nullptr, nullptr, 0, nullptr,
             t_de1, dec_b1, t_de2, dec_b2, nullptr, nullptr, nullptr, dec_h);
        sitl_dec3<<<gN, B256, 0, stream>>>(dec_h, dec_w3, dec_b3, gns);

        // ---- semi-implicit integration ------------------------------------
        sitl_integrate<<<gN3, B256, 0, stream>>>(u, r, acc_sph, gns);
    }

    sitl_final<<<gN3, B256, 0, stream>>>(r, r0, u0, (float*)d_out);
}
